// BANDNN_90082644067049
// MI455X (gfx1250) — matrix operand order repaired and checked by reading, not compiled, not run on hardware
//
#include <hip/hip_runtime.h>
#include <hip/hip_bf16.h>

// ---------------------------------------------------------------------------
// BANDNN energy MLP on CDNA5 (gfx1250), wave32 + v_wmma_f32_16x16x32_bf16.
//
// Per edge type: rows X (N x in_dim) -> MLP (in->128->256->128->1, ReLU) ->
// segment_sum into out[512]. Compute-bound (~40 GFLOP vs ~12 MB traffic),
// so everything is routed through BF16 WMMA with fp32 accumulation.
// ---------------------------------------------------------------------------

typedef __attribute__((ext_vector_type(16))) __bf16        v16bf;
typedef __attribute__((ext_vector_type(8)))  float         v8f;
typedef __attribute__((ext_vector_type(8)))  unsigned int  v8u;

// ---- A fragment: 16x32 bf16, rows in LDS, row-major with stride kdim ------
// Layout (ISA 7.12.2, 16-bit A 16x32): lanes 0-15 hold row m=lane,
// VGPR j<4 -> K=2j..2j+1, VGPR j>=4 -> K=16+2(j-4)..; lanes 16-31 are K+8.
__device__ __forceinline__ v16bf load_a_frag(const __bf16* base, int kdim, int k0) {
    const int l    = threadIdx.x & 31;
    const int half = l >> 4;
    const int m    = l & 15;
    const unsigned int* p =
        (const unsigned int*)base + ((m * kdim + k0) >> 1) + half * 4;
    v8u t;
    t[0] = p[0]; t[1] = p[1]; t[2] = p[2]; t[3] = p[3];   // K = {0..7}  (+8*half)
    t[4] = p[8]; t[5] = p[9]; t[6] = p[10]; t[7] = p[11]; // K = {16..23}(+8*half)
    return __builtin_bit_cast(v16bf, t);
}

// ---- B fragment: 32x16 bf16 from global weights stored [N][Kpad] ----------
// lane n<16 holds column n, K=0..7,16..23 (pairs per VGPR); lanes 16-31 K+8.
// [N][K] storage gives each lane's 8 dwords as two contiguous 16B runs.
__device__ __forceinline__ v16bf load_b_frag(const __bf16* __restrict__ w,
                                             int kdim, int n0, int k0) {
    const int l    = threadIdx.x & 31;
    const int half = l >> 4;
    const int n    = n0 + (l & 15);
    const unsigned int* p =
        (const unsigned int*)w + ((n * kdim + k0 + 8 * half) >> 1);
    v8u t;
#pragma unroll
    for (int j = 0; j < 8; ++j) t[j] = p[(j & 3) + 8 * (j >> 2)];
    return __builtin_bit_cast(v16bf, t);
}

// ---- C tile (16x16 f32, 8 VGPRs) -> bias + ReLU -> bf16 LDS row-major -----
__device__ __forceinline__ void store_c_relu(__bf16* sOut, int ndim, int n0,
                                             const v8f& c, float bias) {
    const int l    = threadIdx.x & 31;
    const int half = l >> 4;
    const int n    = n0 + (l & 15);
#pragma unroll
    for (int r = 0; r < 8; ++r) {
        const int m = r + 8 * half;
        float v = fmaxf(c[r] + bias, 0.0f);
        sOut[m * ndim + n] = (__bf16)v;
    }
}

// ---- One dense layer over 4 M-tiles (64 rows) -----------------------------
// B fragments are loaded exactly once per (nt,ks) and reused across 4 M-tiles.
template <int KDIM, int NDIM>
__device__ __forceinline__ void mlp_layer(const __bf16* sIn, __bf16* sOut,
                                          const __bf16* __restrict__ wt,
                                          const float* __restrict__ bias) {
    const int wave  = threadIdx.x >> 5;
    const int nlane = threadIdx.x & 15;
    for (int nt = wave; nt < (NDIM >> 4); nt += 4) {
        v8f c[4];
#pragma unroll
        for (int mt = 0; mt < 4; ++mt) c[mt] = v8f{};
#pragma unroll
        for (int ks = 0; ks < KDIM; ks += 32) {
            v16bf b = load_b_frag(wt, KDIM, nt * 16, ks);
#pragma unroll
            for (int mt = 0; mt < 4; ++mt) {
                v16bf a = load_a_frag(sIn + mt * 16 * KDIM, KDIM, ks);
                c[mt] = __builtin_amdgcn_wmma_f32_16x16x32_bf16(
                    false, a, false, b, (short)0, c[mt], false, false);
            }
        }
        const float bs = bias[nt * 16 + nlane];
#pragma unroll
        for (int mt = 0; mt < 4; ++mt)
            store_c_relu(sOut + mt * 16 * NDIM, NDIM, nt * 16, c[mt], bs);
    }
}

// ---- Main fused MLP + segment-sum kernel: 64 rows / workgroup -------------
__global__ __launch_bounds__(128) void bandnn_mlp(
    const float* __restrict__ X, const int* __restrict__ seg,
    const __bf16* __restrict__ wts,  // [N][Kpad] bf16: W1t | W2t | W3t
    const float* __restrict__ b1, const float* __restrict__ b2,
    const float* __restrict__ b3, const float* __restrict__ w4,
    const float* __restrict__ b4, int in_dim, float* __restrict__ out) {
    __shared__ alignas(16) __bf16 sX [64 * 32];   //  4 KB
    __shared__ alignas(16) __bf16 sHa[64 * 128];  // 16 KB
    __shared__ alignas(16) __bf16 sHb[64 * 256];  // 32 KB
    const int  tid = threadIdx.x;
    const long r0  = (long)blockIdx.x * 64;

    // Stage X tile (64 x in_dim) -> bf16, zero-padded to K=32.
    for (int e = tid; e < 64 * 32; e += 128) {
        const int row = e >> 5, col = e & 31;
        float v = (col < in_dim) ? X[(r0 + row) * in_dim + col] : 0.0f;
        sX[e] = (__bf16)v;
    }
    __syncthreads();

    mlp_layer<32, 128>(sX, sHa, wts, b1);             // layer 1
    __syncthreads();
    mlp_layer<128, 256>(sHa, sHb, wts + 4096, b2);    // layer 2
    __syncthreads();
    mlp_layer<256, 128>(sHb, sHa, wts + 36864, b3);   // layer 3
    __syncthreads();

    // Layer 4: (64x128) @ (128x1) + b4, then per-molecule atomic scatter.
    {
        const int row  = tid >> 1;          // 2 lanes per row, same wave
        const int part = tid & 1;
        const __bf16* h = sHa + row * 128 + part * 64;
        const float*  w = w4 + part * 64;
        float s = 0.0f;
#pragma unroll
        for (int i = 0; i < 64; ++i) s += (float)h[i] * w[i];
        s += __shfl_xor(s, 1);
        if (part == 0)
            atomicAdd(out + seg[r0 + row], s + b4[0]);
    }
}

// ---- Weight pre-pass: fp32 [K][N] -> bf16 [N][Kpad], W1 K padded to 32 ----
__global__ void convert_weights(const float* __restrict__ w1,
                                const float* __restrict__ w2,
                                const float* __restrict__ w3,
                                int in_dim, __bf16* __restrict__ outw) {
    const int i = blockIdx.x * blockDim.x + threadIdx.x;
    if (i < 4096) {                       // W1t: 128 x 32
        const int n = i >> 5, k = i & 31;
        float v = (k < in_dim) ? w1[k * 128 + n] : 0.0f;
        outw[n * 32 + k] = (__bf16)v;
    } else if (i < 4096 + 32768) {        // W2t: 256 x 128
        const int j = i - 4096;
        const int n = j >> 7, k = j & 127;
        outw[4096 + n * 128 + k] = (__bf16)w2[k * 256 + n];
    } else if (i < 69632) {               // W3t: 128 x 256
        const int j = i - 36864;
        const int n = j >> 8, k = j & 255;
        outw[36864 + n * 256 + k] = (__bf16)w3[k * 128 + n];
    }
}

__global__ void zero_out(float* __restrict__ out, int n) {
    const int i = blockIdx.x * blockDim.x + threadIdx.x;
    if (i < n) out[i] = 0.0f;
}

// ---------------------------------------------------------------------------
extern "C" void kernel_launch(void* const* d_in, const int* in_sizes, int n_in,
                              void* d_out, int out_size, void* d_ws, size_t ws_size,
                              hipStream_t stream) {
    const float* xs[4]   = {(const float*)d_in[0], (const float*)d_in[2],
                            (const float*)d_in[4], (const float*)d_in[6]};
    const int*   segs[4] = {(const int*)d_in[1], (const int*)d_in[3],
                            (const int*)d_in[5], (const int*)d_in[7]};
    const int in_dims[4] = {13, 19, 6, 25};
    const int nrows[4]   = {15360, 25600, 204800, 40960};

    // Params pytree may arrive either as 8 leaves per type (n_in == 40) or
    // as one flat concatenated buffer per type (n_in == 12).
    const float* W[4][4];
    const float* Bv[4][4];
    if (n_in >= 40) {
        for (int t = 0; t < 4; ++t)
            for (int j = 0; j < 4; ++j) {
                W[t][j]  = (const float*)d_in[8 + t * 8 + 2 * j];
                Bv[t][j] = (const float*)d_in[8 + t * 8 + 2 * j + 1];
            }
    } else {
        for (int t = 0; t < 4; ++t) {
            const float* p = (const float*)d_in[8 + t];
            const int din = in_dims[t];
            W[t][0]  = p;  p += din * 128;
            Bv[t][0] = p;  p += 128;
            W[t][1]  = p;  p += 128 * 256;
            Bv[t][1] = p;  p += 256;
            W[t][2]  = p;  p += 256 * 128;
            Bv[t][2] = p;  p += 128;
            W[t][3]  = p;  p += 128;
            Bv[t][3] = p;
        }
    }

    __bf16* wbf = (__bf16*)d_ws;  // 4 * 69632 bf16 = 557,056 bytes of scratch
    for (int t = 0; t < 4; ++t)
        convert_weights<<<(69632 + 255) / 256, 256, 0, stream>>>(
            W[t][0], W[t][1], W[t][2], in_dims[t], wbf + t * 69632);

    zero_out<<<(out_size + 255) / 256, 256, 0, stream>>>((float*)d_out, out_size);

    for (int t = 0; t < 4; ++t)
        bandnn_mlp<<<nrows[t] / 64, 128, 0, stream>>>(
            xs[t], segs[t], wbf + t * 69632, Bv[t][0], Bv[t][1], Bv[t][2],
            W[t][3], Bv[t][3], in_dims[t], (float*)d_out);
}
